// ManBlock_6837587935728
// MI455X (gfx1250) — matrix operand order repaired and checked by reading, not compiled, not run on hardware
//
#include <hip/hip_runtime.h>
#include <math.h>

#define B_ 4
#define C_ 32
#define H_ 128
#define W_ 128
#define NH_ 4
#define DK_ 8
#define HW_ (H_*W_)
#define SLOT 2097152  // floats per workspace slot (8 MB)

typedef _Float16 h16v __attribute__((ext_vector_type(16)));
typedef float    f8v  __attribute__((ext_vector_type(8)));

// ---- WMMA fragment helpers (gfx1250 v_wmma_f32_16x16x32_f16) ----
// A (16x32, MxK) f16, row-major in LDS with row stride `stride` halfs:
// lane L: m = L%16, kb = (L/16)*8; elems 0..7 = A[m][kb..kb+7], 8..15 = A[m][16+kb..16+kb+7]
__device__ __forceinline__ h16v load_a(const _Float16* base, int stride, int lane){
  const _Float16* p = base + (lane & 15) * stride + ((lane >> 4) << 3);
  h16v a;
#pragma unroll
  for (int i = 0; i < 8; ++i){ a[i] = p[i]; a[8 + i] = p[16 + i]; }
  return a;
}
// B (32x16, KxN) f16 from *transposed* LDS Bt[n][k] (row stride `stride`):
// lane L: n = L%16, kb = (L/16)*8; elems 0..7 = Bt[n][kb..kb+7], 8..15 = Bt[n][16+kb..16+kb+7]
__device__ __forceinline__ h16v load_bt(const _Float16* baseT, int stride, int lane){
  const _Float16* p = baseT + (lane & 15) * stride + ((lane >> 4) << 3);
  h16v b;
#pragma unroll
  for (int i = 0; i < 8; ++i){ b[i] = p[i]; b[8 + i] = p[16 + i]; }
  return b;
}
__device__ __forceinline__ f8v wmma16(h16v a, h16v b, f8v c){
  return __builtin_amdgcn_wmma_f32_16x16x32_f16(false, a, false, b, (short)0, c, false, false);
}
// C/D layout: element r of lane L -> M = r + 8*(L/16), N = L%16

// ---------------- K1: bilinear resize (align-corners) 32x32 -> 128x128 ----------------
__global__ void k_resize(const float* __restrict__ fg, float* __restrict__ out){
  int t = blockIdx.x * blockDim.x + threadIdx.x;
  if (t >= B_*C_*HW_) return;
  int x = t % W_, y = (t / W_) % H_, c = (t / HW_) % C_, b = t / (C_*HW_);
  const float sc = 31.0f / 127.0f;
  float ys = y * sc, xs = x * sc;
  int y0 = (int)floorf(ys); int y1 = (y0 + 1 < 31) ? y0 + 1 : 31;
  int x0 = (int)floorf(xs); int x1 = (x0 + 1 < 31) ? x0 + 1 : 31;
  float wy = ys - (float)y0, wx = xs - (float)x0;
  const float* p = fg + ((size_t)(b*C_ + c)) * 1024;
  float v00 = p[y0*32+x0], v01 = p[y0*32+x1], v10 = p[y1*32+x0], v11 = p[y1*32+x1];
  float r0 = v00*(1.0f-wx) + v01*wx, r1 = v10*(1.0f-wx) + v11*wx;
  out[t] = r0*(1.0f-wy) + r1*wy;
}

// ---------------- K2: fused QKV projection + theta_shift (WMMA) ----------------
// one block per (b,h) image row; 8 waves x 16-pixel M-tiles; N = 96 (q|k|v), K = 32
__global__ void __launch_bounds__(256) k_qkv(
    const float* __restrict__ x,
    const float* __restrict__ wq, const float* __restrict__ bq,
    const float* __restrict__ wk, const float* __restrict__ bk,
    const float* __restrict__ wv, const float* __restrict__ bv,
    float* __restrict__ qr, float* __restrict__ kr, float* __restrict__ vO){
  __shared__ _Float16 Ax[128*40];
  __shared__ _Float16 Bw[96*40];
  int tid = threadIdx.x, blk = blockIdx.x;
  int b = blk / H_, h = blk % H_;
  for (int e = tid; e < 128*C_; e += 256){
    int w = e & 127, c = e >> 7;
    Ax[w*40 + c] = (_Float16)x[((size_t)(b*C_ + c)*H_ + h)*W_ + w];
  }
  for (int e = tid; e < 96*C_; e += 256){
    int n = e / C_, k = e % C_;
    float val = (n < 32) ? wq[n*C_+k] : ((n < 64) ? wk[(n-32)*C_+k] : wv[(n-64)*C_+k]);
    Bw[n*40 + k] = (_Float16)val;
  }
  __syncthreads();
  int wave = tid >> 5, lane = tid & 31;
  h16v a = load_a(Ax + (wave*16)*40, 40, lane);
  const float kscale = 0.35355339059327379f; // DK^-0.5
#pragma unroll
  for (int nt = 0; nt < 6; ++nt){
    h16v bf = load_bt(Bw + (nt*16)*40, 40, lane);
    f8v acc = {};
    acc = wmma16(a, bf, acc);
    int n96 = nt*16 + (lane & 15);
#pragma unroll
    for (int r = 0; r < 8; ++r){
      int pw = wave*16 + r + 8*(lane >> 4);
      float val = acc[r];
      if (n96 < 32)       val += bq[n96];
      else if (n96 < 64)  val = (val + bk[n96-32]) * kscale;
      else                val += bv[n96-64];
      if (n96 < 64){ // uniform per nt
        float partner = __shfl_xor(val, 1, 32);          // paired rotary channel
        float rot = (n96 & 1) ? partner : -partner;      // even: -x_odd ; odd: +x_even
        int nlocal = n96 & 31, dk = nlocal & 7;
        float ang = __expf(-3.0701134573253944f * (float)(dk >> 1)); // 10000^{-j/3}
        float ph = (float)(h*W_ + pw) * ang;
        float s, co; __sincosf(ph, &s, &co);
        val = val*co + rot*s;
        int nh = nlocal >> 3;
        float* dst = (n96 < 32) ? qr : kr;
        dst[(((size_t)(b*NH_ + nh)*H_ + h)*W_ + pw)*DK_ + dk] = val;
      } else {
        int c = n96 - 64;
        vO[((size_t)(b*C_ + c)*H_ + h)*W_ + pw] = val;
      }
    }
  }
}

// ---------------- K3: depthwise 5x5 LePE conv ----------------
__global__ void k_lepe(const float* __restrict__ v, const float* __restrict__ lw,
                       const float* __restrict__ lb, float* __restrict__ out){
  int t = blockIdx.x * blockDim.x + threadIdx.x;
  if (t >= B_*C_*HW_) return;
  int w = t % W_, h = (t / W_) % H_, c = (t / HW_) % C_, b = t / (C_*HW_);
  const float* vp = v + ((size_t)(b*C_ + c)) * HW_;
  const float* wp = lw + c*25;
  float acc = lb[c];
#pragma unroll
  for (int dy = 0; dy < 5; ++dy){
    int yy = h + dy - 2; if (yy < 0 || yy >= H_) continue;
#pragma unroll
    for (int dx = 0; dx < 5; ++dx){
      int xx = w + dx - 2; if (xx < 0 || xx >= W_) continue;
      acc += vp[yy*W_ + xx] * wp[dy*5 + dx];
    }
  }
  out[t] = acc;
}

// ---------------- K4: row retention: softmax(QK^T + mask_w) @ V -> v1 ----------------
// one block per (b,n,h). S is 128x128, kept in registers (8 WMMA C-tiles / wave).
__global__ void __launch_bounds__(256) k_attn_w(
    const float* __restrict__ qr, const float* __restrict__ kr,
    const float* __restrict__ v, float* __restrict__ v1){
  __shared__ _Float16 Qs[128*36];
  __shared__ _Float16 Ks[128*36];
  __shared__ _Float16 Vd[16*132];
  __shared__ _Float16 Ps[8*16*132];
  int tid = threadIdx.x, blk = blockIdx.x;
  int b = blk / (NH_*H_); int rem = blk % (NH_*H_); int n = rem / H_, h = rem % H_;
  size_t qbase = (((size_t)(b*NH_ + n)*H_ + h) * W_) * DK_;
  for (int e = tid; e < 128*36; e += 256){
    int i = e / 36, k = e % 36;
    _Float16 zq = (_Float16)0.0f, zk = (_Float16)0.0f;
    if (k < DK_){ zq = (_Float16)qr[qbase + i*DK_ + k]; zk = (_Float16)kr[qbase + i*DK_ + k]; }
    Qs[e] = zq; Ks[e] = zk;
  }
  for (int e = tid; e < 16*132; e += 256){
    int d = e / 132, j = e % 132;
    float val = (d < DK_ && j < W_) ? v[((size_t)(b*C_ + n*DK_ + d)*H_ + h)*W_ + j] : 0.0f;
    Vd[e] = (_Float16)val;
  }
  __syncthreads();
  int wave = tid >> 5, lane = tid & 31;
  int halfsel = lane >> 4, nlo = lane & 15;
  float decay = logf(1.0f - exp2f(-1.0f - 0.75f*(float)n));
  h16v aq = load_a(Qs + (wave*16)*36, 36, lane);
  f8v scm[8];
#pragma unroll
  for (int jt = 0; jt < 8; ++jt){
    h16v bfr = load_bt(Ks + (jt*16)*36, 36, lane);
    f8v z = {};
    scm[jt] = wmma16(aq, bfr, z);
  }
  float rmax[8], rsum[8];
#pragma unroll
  for (int r = 0; r < 8; ++r){
    int i = wave*16 + r + 8*halfsel;
    float m = -3.0e38f;
#pragma unroll
    for (int jt = 0; jt < 8; ++jt){
      int j = jt*16 + nlo;
      scm[jt][r] += decay * fabsf((float)(i - j));
      m = fmaxf(m, scm[jt][r]);
    }
    rmax[r] = m;
  }
#pragma unroll
  for (int off = 1; off < 16; off <<= 1)
#pragma unroll
    for (int r = 0; r < 8; ++r) rmax[r] = fmaxf(rmax[r], __shfl_xor(rmax[r], off, 32));
#pragma unroll
  for (int r = 0; r < 8; ++r){
    float s = 0.0f;
#pragma unroll
    for (int jt = 0; jt < 8; ++jt){ scm[jt][r] = __expf(scm[jt][r] - rmax[r]); s += scm[jt][r]; }
    rsum[r] = s;
  }
#pragma unroll
  for (int off = 1; off < 16; off <<= 1)
#pragma unroll
    for (int r = 0; r < 8; ++r) rsum[r] += __shfl_xor(rsum[r], off, 32);
  _Float16* Pw = Ps + wave*16*132;
#pragma unroll
  for (int r = 0; r < 8; ++r){
    float inv = 1.0f / rsum[r];
    int il = r + 8*halfsel;
#pragma unroll
    for (int jt = 0; jt < 8; ++jt) Pw[il*132 + jt*16 + nlo] = (_Float16)(scm[jt][r] * inv);
  }
  __syncthreads();
  f8v oacc = {};
#pragma unroll
  for (int ks = 0; ks < 4; ++ks){
    h16v ap = load_a(Pw + ks*32, 132, lane);
    h16v bp = load_bt(Vd + ks*32, 132, lane);
    oacc = wmma16(ap, bp, oacc);
  }
  if (nlo < DK_){
#pragma unroll
    for (int r = 0; r < 8; ++r){
      int i = wave*16 + r + 8*halfsel;
      v1[qbase + i*DK_ + nlo] = oacc[r];
    }
  }
}

// ---------------- K5: column retention: softmax(QK^T + mask_h) @ v1 -> attn (BHWC) ----------------
__global__ void __launch_bounds__(256) k_attn_h(
    const float* __restrict__ qr, const float* __restrict__ kr,
    const float* __restrict__ v1, float* __restrict__ attn){
  __shared__ _Float16 Qs[128*36];
  __shared__ _Float16 Ks[128*36];
  __shared__ _Float16 Vd[16*132];
  __shared__ _Float16 Ps[8*16*132];
  int tid = threadIdx.x, blk = blockIdx.x;
  int b = blk / (NH_*W_); int rem = blk % (NH_*W_); int n = rem / W_, w = rem % W_;
  size_t base = ((size_t)(b*NH_ + n)) * H_ * W_ * DK_ + (size_t)w * DK_;
  for (int e = tid; e < 128*36; e += 256){
    int i = e / 36, k = e % 36;
    _Float16 zq = (_Float16)0.0f, zk = (_Float16)0.0f;
    if (k < DK_){
      zq = (_Float16)qr[base + (size_t)i*W_*DK_ + k];
      zk = (_Float16)kr[base + (size_t)i*W_*DK_ + k];
    }
    Qs[e] = zq; Ks[e] = zk;
  }
  for (int e = tid; e < 16*132; e += 256){
    int d = e / 132, j = e % 132;
    float val = (d < DK_ && j < H_) ? v1[base + (size_t)j*W_*DK_ + d] : 0.0f;
    Vd[e] = (_Float16)val;
  }
  __syncthreads();
  int wave = tid >> 5, lane = tid & 31;
  int halfsel = lane >> 4, nlo = lane & 15;
  float decay = logf(1.0f - exp2f(-1.0f - 0.75f*(float)n));
  h16v aq = load_a(Qs + (wave*16)*36, 36, lane);
  f8v scm[8];
#pragma unroll
  for (int jt = 0; jt < 8; ++jt){
    h16v bfr = load_bt(Ks + (jt*16)*36, 36, lane);
    f8v z = {};
    scm[jt] = wmma16(aq, bfr, z);
  }
  float rmax[8], rsum[8];
#pragma unroll
  for (int r = 0; r < 8; ++r){
    int i = wave*16 + r + 8*halfsel;
    float m = -3.0e38f;
#pragma unroll
    for (int jt = 0; jt < 8; ++jt){
      int j = jt*16 + nlo;
      scm[jt][r] += decay * fabsf((float)(i - j));
      m = fmaxf(m, scm[jt][r]);
    }
    rmax[r] = m;
  }
#pragma unroll
  for (int off = 1; off < 16; off <<= 1)
#pragma unroll
    for (int r = 0; r < 8; ++r) rmax[r] = fmaxf(rmax[r], __shfl_xor(rmax[r], off, 32));
#pragma unroll
  for (int r = 0; r < 8; ++r){
    float s = 0.0f;
#pragma unroll
    for (int jt = 0; jt < 8; ++jt){ scm[jt][r] = __expf(scm[jt][r] - rmax[r]); s += scm[jt][r]; }
    rsum[r] = s;
  }
#pragma unroll
  for (int off = 1; off < 16; off <<= 1)
#pragma unroll
    for (int r = 0; r < 8; ++r) rsum[r] += __shfl_xor(rsum[r], off, 32);
  _Float16* Pw = Ps + wave*16*132;
#pragma unroll
  for (int r = 0; r < 8; ++r){
    float inv = 1.0f / rsum[r];
    int il = r + 8*halfsel;
#pragma unroll
    for (int jt = 0; jt < 8; ++jt) Pw[il*132 + jt*16 + nlo] = (_Float16)(scm[jt][r] * inv);
  }
  __syncthreads();
  f8v oacc = {};
#pragma unroll
  for (int ks = 0; ks < 4; ++ks){
    h16v ap = load_a(Pw + ks*32, 132, lane);
    h16v bp = load_bt(Vd + ks*32, 132, lane);
    oacc = wmma16(ap, bp, oacc);
  }
  if (nlo < DK_){
#pragma unroll
    for (int r = 0; r < 8; ++r){
      int i = wave*16 + r + 8*halfsel;
      attn[((size_t)(b*H_ + i)*W_ + w)*C_ + n*DK_ + nlo] = oacc[r];
    }
  }
}

// ---------------- K6: (attn + lepe) @ wo.T + bo -> at (NCHW, WMMA) ----------------
__global__ void __launch_bounds__(256) k_outproj(
    const float* __restrict__ attn, const float* __restrict__ lepe,
    const float* __restrict__ wo, const float* __restrict__ bo,
    float* __restrict__ at){
  __shared__ _Float16 Ax[128*40];
  __shared__ _Float16 Bw[32*40];
  int tid = threadIdx.x, blk = blockIdx.x;
  int b = blk / H_, h = blk % H_;
  for (int e = tid; e < 128*C_; e += 256){          // attn BHWC (c fastest)
    int c = e & 31, w = e >> 5;
    Ax[w*40 + c] = (_Float16)attn[((size_t)(b*H_ + h)*W_ + w)*C_ + c];
  }
  __syncthreads();
  for (int e = tid; e < 128*C_; e += 256){          // + lepe NCHW (w fastest)
    int w = e & 127, c = e >> 7;
    float add = lepe[((size_t)(b*C_ + c)*H_ + h)*W_ + w];
    Ax[w*40 + c] = (_Float16)((float)Ax[w*40 + c] + add);
  }
  for (int e = tid; e < 32*C_; e += 256){
    int n = e / C_, k = e % C_;
    Bw[n*40 + k] = (_Float16)wo[n*C_ + k];
  }
  __syncthreads();
  int wave = tid >> 5, lane = tid & 31;
  h16v a = load_a(Ax + wave*16*40, 40, lane);
#pragma unroll
  for (int nt = 0; nt < 2; ++nt){
    h16v bf = load_bt(Bw + nt*16*40, 40, lane);
    f8v acc = {};
    acc = wmma16(a, bf, acc);
    int nch = nt*16 + (lane & 15);
#pragma unroll
    for (int r = 0; r < 8; ++r){
      int pw = wave*16 + r + 8*(lane >> 4);
      at[((size_t)(b*C_ + nch)*H_ + h)*W_ + pw] = acc[r] + bo[nch];
    }
  }
}

// ---------------- K7: per-(b,c) mean & max over HxW ----------------
__global__ void __launch_bounds__(256) k_pool(const float* __restrict__ at,
    float* __restrict__ meanbc, float* __restrict__ maxbc){
  __shared__ float sm[256], sx[256];
  int bc = blockIdx.x;
  const float* p = at + (size_t)bc * HW_;
  float s = 0.0f, m = -3.0e38f;
  for (int i = threadIdx.x; i < HW_; i += 256){ float v = p[i]; s += v; m = fmaxf(m, v); }
  sm[threadIdx.x] = s; sx[threadIdx.x] = m;
  __syncthreads();
  for (int off = 128; off > 0; off >>= 1){
    if (threadIdx.x < (unsigned)off){
      sm[threadIdx.x] += sm[threadIdx.x + off];
      sx[threadIdx.x] = fmaxf(sx[threadIdx.x], sx[threadIdx.x + off]);
    }
    __syncthreads();
  }
  if (threadIdx.x == 0){ meanbc[bc] = sm[0] / (float)HW_; maxbc[bc] = sx[0]; }
}

// ---------------- K8: tiny gating heads (at_r, att_c) ----------------
__global__ void __launch_bounds__(128) k_small(
    const float* __restrict__ meanbc, const float* __restrict__ maxbc,
    const float* __restrict__ fb_w1, const float* __restrict__ g1, const float* __restrict__ b1,
    const float* __restrict__ g2, const float* __restrict__ b2,
    const float* __restrict__ fb_w2, const float* __restrict__ fb_b2,
    const float* __restrict__ ca_w1, const float* __restrict__ ca_w2,
    float* __restrict__ at_r, float* __restrict__ att_c){
  __shared__ float ts[4][32];
  int tid = threadIdx.x; int b = tid >> 5, co = tid & 31;
  const float rn = rsqrtf(1.0f + 1e-5f);
  float t1 = 0.0f;
  for (int ci = 0; ci < 32; ++ci) t1 += meanbc[b*32 + ci] * fb_w1[(co*32 + ci)*9 + 4]; // 3x3 on 1x1: center tap
  float t2 = t1 * (g1[co]*rn) + b1[co];
  float t3 = fmaxf(0.0f, t2 * (g2[co]*rn) + b2[co]);
  ts[b][co] = t3;
  __syncthreads();
  float z = fb_b2[co];
  for (int ci = 0; ci < 32; ++ci) z += ts[b][ci] * fb_w2[co*32 + ci];
  at_r[tid] = 1.0f / (1.0f + __expf(-z));
  float hm0 = 0, hm1 = 0, hx0 = 0, hx1 = 0;
  for (int c = 0; c < 32; ++c){
    hm0 += ca_w1[c]      * meanbc[b*32 + c];
    hm1 += ca_w1[32 + c] * meanbc[b*32 + c];
    hx0 += ca_w1[c]      * maxbc[b*32 + c];
    hx1 += ca_w1[32 + c] * maxbc[b*32 + c];
  }
  hm0 = fmaxf(hm0, 0.0f); hm1 = fmaxf(hm1, 0.0f);
  hx0 = fmaxf(hx0, 0.0f); hx1 = fmaxf(hx1, 0.0f);
  float zm = ca_w2[co*2]*hm0 + ca_w2[co*2+1]*hm1;
  float zx = ca_w2[co*2]*hx0 + ca_w2[co*2+1]*hx1;
  att_c[tid] = 1.0f / (1.0f + __expf(-(zm + zx)));
}

// ---------------- K9: fused c1/c2 gated GEMMs + fu GEMM + BN + relu -> ff (BHWC) ----------------
__global__ void __launch_bounds__(256) k_fuse(
    const float* __restrict__ f_m, const float* __restrict__ fg_rs,
    const float* __restrict__ at_r,
    const float* __restrict__ c1w, const float* __restrict__ c1b,
    const float* __restrict__ c2w, const float* __restrict__ c2b,
    const float* __restrict__ fuw, const float* __restrict__ fub,
    const float* __restrict__ bng, const float* __restrict__ bnb,
    float* __restrict__ ff){
  __shared__ _Float16 Am[128*40];
  __shared__ _Float16 Ag[128*40];
  __shared__ _Float16 A2[128*72];
  __shared__ _Float16 B1[32*40];
  __shared__ _Float16 B2w[32*40];
  __shared__ _Float16 Bf[32*72];
  __shared__ float invr[32], atr_s[32];
  int tid = threadIdx.x, blk = blockIdx.x;
  int b = blk / H_, h = blk % H_;
  if (tid < 32){ float a = at_r[b*32 + tid]; atr_s[tid] = a; invr[tid] = 1.0f / a; }
  __syncthreads();
  for (int e = tid; e < 128*C_; e += 256){
    int w = e & 127, c = e >> 7;
    float sc = atr_s[c];
    Am[w*40 + c] = (_Float16)(f_m [((size_t)(b*C_+c)*H_+h)*W_ + w] * sc);
    Ag[w*40 + c] = (_Float16)(fg_rs[((size_t)(b*C_+c)*H_+h)*W_ + w] * sc);
  }
  for (int e = tid; e < 32*C_; e += 256){
    int n = e / C_, k = e % C_;
    B1 [n*40 + k] = (_Float16)c1w[n*C_ + k];
    B2w[n*40 + k] = (_Float16)c2w[n*C_ + k];
  }
  for (int e = tid; e < 32*64; e += 256){
    int n = e >> 6, k = e & 63;
    Bf[n*72 + k] = (_Float16)fuw[n*64 + k];
  }
  __syncthreads();
  int wave = tid >> 5, lane = tid & 31;
  h16v am = load_a(Am + wave*16*40, 40, lane);
  h16v ag = load_a(Ag + wave*16*40, 40, lane);
#pragma unroll
  for (int nt = 0; nt < 2; ++nt){
    h16v b1f = load_bt(B1  + nt*16*40, 40, lane);
    h16v b2f = load_bt(B2w + nt*16*40, 40, lane);
    f8v d1 = {}; d1 = wmma16(am, b1f, d1);
    f8v d2 = {}; d2 = wmma16(ag, b2f, d2);
    int nch = nt*16 + (lane & 15);
#pragma unroll
    for (int r = 0; r < 8; ++r){
      int pw = wave*16 + r + 8*(lane >> 4);
      float rm = (float)Am[pw*40 + nch] * invr[nch];   // recover ungated residual
      float rg = (float)Ag[pw*40 + nch] * invr[nch];
      A2[pw*72 + nch]      = (_Float16)(d1[r] + c1b[nch] + rm);
      A2[pw*72 + 32 + nch] = (_Float16)(d2[r] + c2b[nch] + rg);
    }
  }
  __syncthreads();
  const float rn = rsqrtf(1.0f + 1e-5f);
#pragma unroll
  for (int nt = 0; nt < 2; ++nt){
    f8v acc = {};
#pragma unroll
    for (int ks = 0; ks < 2; ++ks){
      h16v a2 = load_a(A2 + wave*16*72 + ks*32, 72, lane);
      h16v bf = load_bt(Bf + nt*16*72 + ks*32, 72, lane);
      acc = wmma16(a2, bf, acc);
    }
    int nch = nt*16 + (lane & 15);
#pragma unroll
    for (int r = 0; r < 8; ++r){
      int pw = wave*16 + r + 8*(lane >> 4);
      float vv = (acc[r] + fub[nch]) * (bng[nch]*rn) + bnb[nch];
      ff[((size_t)(b*H_+h)*W_ + pw)*C_ + nch] = fmaxf(vv, 0.0f);
    }
  }
}

// ---------------- K10a: channel mean/max of at*att_c per pixel ----------------
__global__ void k_sa_pool(const float* __restrict__ at, const float* __restrict__ att_c,
                          float* __restrict__ smean, float* __restrict__ smax){
  int t = blockIdx.x * blockDim.x + threadIdx.x;
  if (t >= B_*HW_) return;
  int b = t / HW_, hw = t % HW_;
  float s = 0.0f, m = -3.0e38f;
  for (int c = 0; c < 32; ++c){
    float v = at[((size_t)(b*C_ + c))*HW_ + hw] * att_c[b*32 + c];
    s += v; m = fmaxf(m, v);
  }
  smean[t] = s / 32.0f; smax[t] = m;
}

// ---------------- K10b: 7x7 conv on [mean,max] -> sigmoid spatial map ----------------
__global__ void k_sa_conv(const float* __restrict__ smean, const float* __restrict__ smax,
                          const float* __restrict__ saw, float* __restrict__ at_sa){
  int t = blockIdx.x * blockDim.x + threadIdx.x;
  if (t >= B_*HW_) return;
  int b = t / HW_, hw = t % HW_; int h = hw / W_, w = hw % W_;
  float acc = 0.0f;
#pragma unroll
  for (int dy = 0; dy < 7; ++dy){
    int yy = h + dy - 3; if (yy < 0 || yy >= H_) continue;
#pragma unroll
    for (int dx = 0; dx < 7; ++dx){
      int xx = w + dx - 3; if (xx < 0 || xx >= W_) continue;
      int off = b*HW_ + yy*W_ + xx;
      acc += smean[off]*saw[dy*7+dx] + smax[off]*saw[49 + dy*7+dx];
    }
  }
  at_sa[t] = 1.0f / (1.0f + __expf(-acc));
}

// ---------------- K11: co 1x1 GEMM on ff + at_ca*at_sa, + residual x ----------------
__global__ void __launch_bounds__(256) k_final(
    const float* __restrict__ ff, const float* __restrict__ at,
    const float* __restrict__ att_c, const float* __restrict__ at_sa,
    const float* __restrict__ cow, const float* __restrict__ cob,
    const float* __restrict__ x, float* __restrict__ out){
  __shared__ _Float16 Ax[128*40];
  __shared__ _Float16 Bw[32*40];
  int tid = threadIdx.x, blk = blockIdx.x;
  int b = blk / H_, h = blk % H_;
  for (int e = tid; e < 128*C_; e += 256){          // ff BHWC
    int c = e & 31, w = e >> 5;
    Ax[w*40 + c] = (_Float16)ff[((size_t)(b*H_+h)*W_ + w)*C_ + c];
  }
  __syncthreads();
  for (int e = tid; e < 128*C_; e += 256){          // + at*att_c*at_sa (NCHW)
    int w = e & 127, c = e >> 7;
    float add = at[((size_t)(b*C_+c)*H_+h)*W_ + w] * att_c[b*32+c] * at_sa[b*HW_ + h*W_ + w];
    Ax[w*40 + c] = (_Float16)((float)Ax[w*40 + c] + add);
  }
  for (int e = tid; e < 32*C_; e += 256){
    int n = e / C_, k = e % C_;
    Bw[n*40 + k] = (_Float16)cow[n*C_ + k];
  }
  __syncthreads();
  int wave = tid >> 5, lane = tid & 31;
  h16v a = load_a(Ax + wave*16*40, 40, lane);
#pragma unroll
  for (int nt = 0; nt < 2; ++nt){
    h16v bf = load_bt(Bw + nt*16*40, 40, lane);
    f8v acc = {};
    acc = wmma16(a, bf, acc);
    int nch = nt*16 + (lane & 15);
#pragma unroll
    for (int r = 0; r < 8; ++r){
      int pw = wave*16 + r + 8*(lane >> 4);
      size_t oi = ((size_t)(b*C_+nch)*H_+h)*W_ + pw;
      out[oi] = acc[r] + cob[nch] + x[oi];
    }
  }
}

extern "C" void kernel_launch(void* const* d_in, const int* in_sizes, int n_in,
                              void* d_out, int out_size, void* d_ws, size_t ws_size,
                              hipStream_t stream) {
  const float* x      = (const float*)d_in[0];
  const float* f_m    = (const float*)d_in[1];
  const float* f_g    = (const float*)d_in[2];
  const float* wq     = (const float*)d_in[3];
  const float* bq     = (const float*)d_in[4];
  const float* wk     = (const float*)d_in[5];
  const float* bk     = (const float*)d_in[6];
  const float* wv     = (const float*)d_in[7];
  const float* bv     = (const float*)d_in[8];
  const float* lepe_w = (const float*)d_in[9];
  const float* lepe_b = (const float*)d_in[10];
  const float* wo     = (const float*)d_in[11];
  const float* bo     = (const float*)d_in[12];
  const float* fb_w1  = (const float*)d_in[13];
  const float* g1     = (const float*)d_in[14];
  const float* b1     = (const float*)d_in[15];
  const float* g2     = (const float*)d_in[16];
  const float* b2     = (const float*)d_in[17];
  const float* fb_w2  = (const float*)d_in[18];
  const float* fb_b2  = (const float*)d_in[19];
  const float* ca_w1  = (const float*)d_in[20];
  const float* ca_w2  = (const float*)d_in[21];
  const float* sa_w   = (const float*)d_in[22];
  const float* fu_w   = (const float*)d_in[23];
  const float* fu_b   = (const float*)d_in[24];
  const float* fu_g   = (const float*)d_in[25];
  const float* fu_bb  = (const float*)d_in[26];
  const float* co_w   = (const float*)d_in[27];
  const float* co_b   = (const float*)d_in[28];
  const float* c1_w   = (const float*)d_in[29];
  const float* c1_b   = (const float*)d_in[30];
  const float* c2_w   = (const float*)d_in[31];
  const float* c2_b   = (const float*)d_in[32];
  float* out = (float*)d_out;

  // Workspace: 6 slots x 8MB = 48MB, lifetime-based reuse.
  float* WS   = (float*)d_ws;
  float* qr   = WS + 0*(size_t)SLOT;
  float* kr   = WS + 1*(size_t)SLOT;
  float* v    = WS + 2*(size_t)SLOT;
  float* lepe = WS + 3*(size_t)SLOT;
  float* v1   = WS + 4*(size_t)SLOT;
  float* fg   = WS + 5*(size_t)SLOT;
  float* attn_pre = v;   // v dead after K4
  float* at       = kr;  // kr dead after K5
  float* ff       = qr;  // qr dead after K5
  float* smean  = WS + 4*(size_t)SLOT;           // v1 dead after K5
  float* smax   = smean + B_*HW_;
  float* at_sa  = smax  + B_*HW_;
  float* meanbc = at_sa + B_*HW_;
  float* maxbc  = meanbc + 128;
  float* at_r   = maxbc  + 128;
  float* att_c  = at_r   + 128;

  const int ETOT = B_*C_*HW_;
  k_resize<<<(ETOT + 255)/256, 256, 0, stream>>>(f_g, fg);
  k_qkv   <<<B_*H_, 256, 0, stream>>>(x, wq, bq, wk, bk, wv, bv, qr, kr, v);
  k_lepe  <<<(ETOT + 255)/256, 256, 0, stream>>>(v, lepe_w, lepe_b, lepe);
  k_attn_w<<<B_*NH_*H_, 256, 0, stream>>>(qr, kr, v, v1);
  k_attn_h<<<B_*NH_*W_, 256, 0, stream>>>(qr, kr, v1, attn_pre);
  k_outproj<<<B_*H_, 256, 0, stream>>>(attn_pre, lepe, wo, bo, at);
  k_pool  <<<B_*C_, 256, 0, stream>>>(at, meanbc, maxbc);
  k_small <<<1, 128, 0, stream>>>(meanbc, maxbc, fb_w1, g1, b1, g2, b2,
                                  fb_w2, fb_b2, ca_w1, ca_w2, at_r, att_c);
  k_fuse  <<<B_*H_, 256, 0, stream>>>(f_m, fg, at_r, c1_w, c1_b, c2_w, c2_b,
                                      fu_w, fu_b, fu_g, fu_bb, ff);
  k_sa_pool<<<(B_*HW_ + 255)/256, 256, 0, stream>>>(at, att_c, smean, smax);
  k_sa_conv<<<(B_*HW_ + 255)/256, 256, 0, stream>>>(smean, smax, sa_w, at_sa);
  k_final <<<B_*H_, 256, 0, stream>>>(ff, at, att_c, at_sa, co_w, co_b, x, out);
}
